// DynamicDilation_65481071395154
// MI455X (gfx1250) — matrix operand order repaired and checked by reading, not compiled, not run on hardware
//
#include <hip/hip_runtime.h>

// ---------------------------------------------------------------------------
// Dynamic-dilation conv bank, fused (CDNA5 / gfx1250, wave32, WMMA f16->f32)
//
// Roofline: 77 GFLOP total (4 levels), 135 MB min traffic -> ~6us @ 23.3TB/s;
// f16 WMMA (16KFLOP/instr) puts compute at a comparable ~15-30us, so we use
// v_wmma_f32_16x16x32_f16 and fuse the per-pixel level select in registers
// (the reference materializes 4x67MB of intermediates + a gather; we don't).
// ---------------------------------------------------------------------------

typedef __attribute__((ext_vector_type(16))) _Float16 v16h;
typedef __attribute__((ext_vector_type(8)))  float    v8f;

#define BATCH 4
#define CIN   64
#define COUT  64
#define Hdim  64
#define Wdim  1024
#define NLEV  4

#define HT    8                 // output rows per block
#define WT    64                // output cols per block
#define HALO  4                 // max dilation
#define RT    (HT + 2*HALO)     // 16 staged rows
#define CT    (WT + 2*HALO)     // 72 staged cols
#define CPAD  68                // channel pad: lane stride 136B = 34 dwords,
                                // gcd(34,64)=2 -> conflict-free strided ds
#define PACKED_ELEMS (NLEV*9*4*2*32*16)   // 147456 f16 weights, packed

// ---------------------------------------------------------------------------
// Pre-pack weights fp32 -> f16 in the exact WMMA A-fragment lane layout
// (ISA 7.12.2, "16-bit A-Matrix 16x32"): lane L holds row M=L&15; lanes 0-15
// carry K in {0..7,16..23}, lanes 16-31 carry K in {8..15,24..31}, 2 K per
// VGPR.  Runtime A-load is then one contiguous 32B read per lane.
// ---------------------------------------------------------------------------
__global__ void dyndil_pack_weights(const float* __restrict__ w,
                                    _Float16* __restrict__ p) {
  int i = blockIdx.x * blockDim.x + threadIdx.x;
  if (i >= PACKED_ELEMS) return;
  int t = i;
  int e    = t & 15;  t >>= 4;   // half index within lane (0..15)
  int lane = t & 31;  t >>= 5;
  int kc   = t & 1;   t >>= 1;   // K chunk (channels 0..31 / 32..63)
  int mt   = t & 3;   t >>= 2;   // M tile (out-channel block of 16)
  int tap  = t % 9;   t /= 9;
  int l    = t;                  // dilation level
  int ky = tap / 3, kx = tap % 3;
  int lo  = (lane < 16);
  int v   = e >> 1, odd = e & 1;
  int k;
  if (v < 4) k = (lo ? 0  : 8)  + 2*v       + odd;
  else       k = (lo ? 16 : 24) + 2*(v - 4) + odd;
  int O = mt*16 + (lane & 15);
  int I = kc*32 + k;
  p[i] = (_Float16)w[(((l*COUT + O)*CIN + I)*3 + ky)*3 + kx];
}

// ---------------------------------------------------------------------------
// Main fused kernel: one block = (b, 8 rows, 64 cols) x all 64 out channels.
// 16 waves: wave = mgroup(2 M-tiles) x row(8).  Per wave, per level: 9 taps x
// 2 K-chunks x (2 A frags reused over 4 N-tiles) -> 144 WMMAs, then per-pixel
// cndmask select against the range-derived dilation index.
// ---------------------------------------------------------------------------
__global__ __launch_bounds__(512) void dyndil_kernel(
    const float* __restrict__ x, const float* __restrict__ range_img,
    const _Float16* __restrict__ wpk, const float* __restrict__ bias,
    float* __restrict__ out)
{
  __shared__ __align__(16) _Float16 tile[RT * CT * CPAD];   // ~153 KB
  __shared__ __align__(16) float    bias_lds[NLEV * COUT];

  const int bx   = blockIdx.x;
  const int wblk = bx & 15;           // Wdim/WT = 16
  const int hblk = (bx >> 4) & 7;     // Hdim/HT = 8
  const int b    = bx >> 7;           // BATCH  = 4
  const int h0   = hblk * HT;
  const int w0   = wblk * WT;
  const int tid  = threadIdx.x;

  if (tid < NLEV * COUT) bias_lds[tid] = bias[tid];

  // ---- stage x tile: fp32 -> f16, channel-contiguous [row][col][c] --------
  const float* xb = x + (size_t)b * CIN * Hdim * Wdim;
  for (int t = tid; t < RT * CT * (CIN/2); t += 512) {
    int col = t % CT;
    int tmp = t / CT;
    int row = tmp % RT;
    int cp  = tmp / RT;                 // channel pair
    int c   = cp * 2;
    int hh  = h0 - HALO + row;
    int ww  = w0 - HALO + col;
    float f0 = 0.0f, f1 = 0.0f;
    if (hh >= 0 && hh < Hdim && ww >= 0 && ww < Wdim) {
      const float* px = xb + ((size_t)c * Hdim + hh) * Wdim + ww;
      f0 = px[0];
      f1 = px[(size_t)Hdim * Wdim];
    }
    union { _Float16 h[2]; unsigned u; } pk;
    pk.h[0] = (_Float16)f0; pk.h[1] = (_Float16)f1;
    int cell = row * CT + col;
    ((unsigned*)tile)[cell * (CPAD/2) + cp] = pk.u;
  }
  __syncthreads();

  // ---- per-wave tile assignment ------------------------------------------
  const int lane = tid & 31;
  const int wid  = tid >> 5;
  const int g    = wid & 7;                 // output row within tile
  const int mb   = (wid >> 3) * 2;          // base M-tile (0 or 2)
  const int ln15 = lane & 15;
  const int hi   = (lane >> 4) & 1;         // lane half (WMMA row split)
  const int h    = h0 + g;

  // per-pixel dilation-level index from range (matches reference math)
  int lvl[4];
#pragma unroll
  for (int nt = 0; nt < 4; ++nt) {
    int wcol = w0 + nt*16 + ln15;
    float r  = range_img[((size_t)b * Hdim + h) * Wdim + wcol];
    r = fminf(fmaxf(r, 5.0f), 50.0f);
    int dil = (int)((50.0f - r) / 11.25f + 1.0f);   // trunc, positive
    dil = min(max(dil, 1), 4);
    lvl[nt] = dil - 1;
  }

  v8f acc[4][2];
#pragma unroll
  for (int nt = 0; nt < 4; ++nt)
#pragma unroll
    for (int mi = 0; mi < 2; ++mi) acc[nt][mi] = (v8f){};

  for (int l = 0; l < NLEV; ++l) {
    const int d = l + 1;

    v8f bj[2];
#pragma unroll
    for (int mi = 0; mi < 2; ++mi)
      bj[mi] = *(const v8f*)(bias_lds + l*COUT + (mb + mi)*16 + hi*8);

    v8f lev[4][2];
#pragma unroll
    for (int nt = 0; nt < 4; ++nt)
#pragma unroll
      for (int mi = 0; mi < 2; ++mi) lev[nt][mi] = (v8f){};

    for (int tap = 0; tap < 9; ++tap) {
      const int ky   = tap / 3, kx = tap % 3;
      const int lrow = HALO + g + (ky - 1) * d;
      const int dcol = (kx - 1) * d;
#pragma unroll
      for (int kc = 0; kc < 2; ++kc) {
        v16h a[2];
#pragma unroll
        for (int mi = 0; mi < 2; ++mi) {
          size_t ai = (((((size_t)l*9 + tap)*4 + (mb + mi))*2 + kc)*32 + lane) * 16;
          a[mi] = *(const v16h*)(wpk + ai);       // 2x global_load_b128, L2 hit
        }
#pragma unroll
        for (int nt = 0; nt < 4; ++nt) {
          int cell = lrow * CT + (HALO + nt*16 + ln15 + dcol);
          const unsigned long long* bp = (const unsigned long long*)
              (tile + cell * CPAD + kc*32 + hi*8);
          union { v16h v; unsigned long long q[4]; } bu;
          bu.q[0] = bp[0]; bu.q[1] = bp[1]; bu.q[2] = bp[4]; bu.q[3] = bp[5];
#pragma unroll
          for (int mi = 0; mi < 2; ++mi)
            lev[nt][mi] = __builtin_amdgcn_wmma_f32_16x16x32_f16(
                false, a[mi], false, bu.v, (short)0, lev[nt][mi], false, false);
        }
      }
    }

    // fused select: keep this level only where the range says so
#pragma unroll
    for (int nt = 0; nt < 4; ++nt) {
      bool sel = (lvl[nt] == l);
#pragma unroll
      for (int mi = 0; mi < 2; ++mi)
#pragma unroll
        for (int j = 0; j < 8; ++j)
          acc[nt][mi][j] = sel ? (lev[nt][mi][j] + bj[mi][j]) : acc[nt][mi][j];
    }
  }

  // ---- store [B][C][H][W] -------------------------------------------------
#pragma unroll
  for (int nt = 0; nt < 4; ++nt) {
    int wcol = w0 + nt*16 + ln15;
#pragma unroll
    for (int mi = 0; mi < 2; ++mi) {
      int Mb = (mb + mi)*16 + hi*8;
#pragma unroll
      for (int j = 0; j < 8; ++j)
        out[(((size_t)b*COUT + (Mb + j))*Hdim + h)*Wdim + wcol] = acc[nt][mi][j];
    }
  }
}

extern "C" void kernel_launch(void* const* d_in, const int* in_sizes, int n_in,
                              void* d_out, int out_size, void* d_ws, size_t ws_size,
                              hipStream_t stream) {
  const float* x    = (const float*)d_in[0];
  const float* rng  = (const float*)d_in[1];
  const float* wts  = (const float*)d_in[2];
  const float* bias = (const float*)d_in[3];
  float* out = (float*)d_out;
  _Float16* wpk = (_Float16*)d_ws;          // needs 294,912 B of scratch

  dyndil_pack_weights<<<(PACKED_ELEMS + 255)/256, 256, 0, stream>>>(wts, wpk);

  dim3 grid(BATCH * (Hdim/HT) * (Wdim/WT));   // 4*8*16 = 512 blocks
  dyndil_kernel<<<grid, 512, 0, stream>>>(x, rng, wpk, bias, out);
}
